// Classifier_1477468749981
// MI455X (gfx1250) — matrix operand order repaired and checked by reading, not compiled, not run on hardware
//
#include <hip/hip_runtime.h>
#include <hip/hip_bf16.h>

typedef __attribute__((ext_vector_type(16))) _Float16 v16h;
typedef __attribute__((ext_vector_type(2)))  _Float16 h2;
typedef __attribute__((ext_vector_type(8)))  float    v8f;

#define SLOPE 0.01f
#define ITERS 30
#define PW    36                  // padded plane width (32 + 2-halo each side)
#define PSZ   (PW * PW)           // 1296 elements per padded plane
#define ZCH   9                   // 8 real channels + 1 zero plane (K-pad, k<=223)
#define HCH   7                   // 6 real channels + 1 zero plane (K-pad, k<=159)

__device__ __forceinline__ float leaky_(float x) { return x >= 0.0f ? x : SLOPE * x; }

// One block per image. 256 threads = 8 wave32 waves. Whole DEQ loop fused in LDS,
// convs as implicit-GEMM via v_wmma_f32_16x16x32_f16 (M padded 6->16, K padded to x32).
// Activations live in zero-halo'd, zero-channel-padded LDS planes so the im2col
// gather is completely branch-free (pure address math + ds_load_u16).
__launch_bounds__(256, 2)
__global__ void deq_fused_kernel(
    const float* __restrict__ image, const float* __restrict__ w1,
    const float* __restrict__ b1,    const float* __restrict__ gamma,
    const float* __restrict__ beta,  const float* __restrict__ w2,
    const float* __restrict__ b2,    const float* __restrict__ wh,
    const float* __restrict__ bh,    float* __restrict__ out)
{
    const int tid   = threadIdx.x;
    const int lane  = tid & 31;
    const int wave  = tid >> 5;
    const int img   = blockIdx.x;
    const int nCol  = lane & 15;     // WMMA column (pixel within tile) / A row
    const int khalf = lane >> 4;     // K half-select for A/B fragments

    __shared__ _Float16 zxp [ZCH * PSZ];   // ch 0-4 = z, 5-7 = image, 8 = zeros (+halo)
    __shared__ _Float16 hpad[HCH * PSZ];   // conv1 activations, ch 6 = zeros (+halo)
    __shared__ _Float16 wa1 [16][224];     // conv1 weights, M 6->16, K 200->224
    __shared__ _Float16 wa2 [16][160];     // conv2 weights, M 5->16, K 150->160
    __shared__ float sb1[6], sb2[5], sg[6], sbt[6];
    __shared__ float gsum[3], gsq[3];
    __shared__ float oacc[10];

    // ---- zero padded planes (halo + zero channels stay zero forever) ----
    for (int i = tid; i < ZCH * PSZ; i += 256) zxp[i]  = (_Float16)0.0f;
    for (int i = tid; i < HCH * PSZ; i += 256) hpad[i] = (_Float16)0.0f;

    // ---- stage weights / params into LDS (f16) ----
    for (int i = tid; i < 16 * 224; i += 256) {
        int m = i / 224, k = i % 224;
        wa1[m][k] = (_Float16)((m < 6 && k < 200) ? w1[m * 200 + k] : 0.0f);
    }
    for (int i = tid; i < 16 * 160; i += 256) {
        int m = i / 160, k = i % 160;
        wa2[m][k] = (_Float16)((m < 5 && k < 150) ? w2[m * 150 + k] : 0.0f);
    }
    if (tid < 6) { sb1[tid] = b1[tid]; sg[tid] = gamma[tid]; sbt[tid] = beta[tid]; }
    if (tid < 5) { sb2[tid] = b2[tid]; }
    if (tid < 3) { gsum[tid] = 0.0f; gsq[tid] = 0.0f; }
    __syncthreads();

    // ---- stage image into interior of planes 5..7 (z planes 0..4 already zero) ----
    const float* ip = image + (size_t)img * 3 * 1024;
    for (int p = tid; p < 1024; p += 256) {
        const int idx = ((p >> 5) + 2) * PW + (p & 31) + 2;
        zxp[5 * PSZ + idx] = (_Float16)ip[p];
        zxp[6 * PSZ + idx] = (_Float16)ip[1024 + p];
        zxp[7 * PSZ + idx] = (_Float16)ip[2048 + p];
    }
    __syncthreads();

    // ================= fixed-point iterations =================
    for (int it = 0; it < ITERS; ++it) {
        // ---------- conv1 (8->6, 5x5 pad 2) + leaky, stats ----------
        float ls[3] = {0.f, 0.f, 0.f}, lq[3] = {0.f, 0.f, 0.f};
        for (int tt = 0; tt < 8; ++tt) {
            const int tile = wave * 8 + tt;
            const int p    = tile * 16 + nCol;
            const int py   = p >> 5, px = p & 31;
            const int base = py * PW + px;          // -2 window offset cancels +2 halo
            v8f acc = {};
            #pragma unroll
            for (int ch = 0; ch < 7; ++ch) {        // K chunks of 32 (224 total)
                const int K0 = ch * 32;
                v16h af, bf;
                #pragma unroll
                for (int v = 0; v < 8; ++v) {       // A fragment: dword pair loads
                    const int kk = K0 + khalf * 16 + 2 * v;
                    const h2 pr = *(const h2*)&wa1[nCol][kk];
                    af[2 * v]     = pr.x;
                    af[2 * v + 1] = pr.y;
                }
                #pragma unroll
                for (int v = 0; v < 8; ++v) {       // B fragment: branch-free gather
                    #pragma unroll
                    for (int j = 0; j < 2; ++j) {
                        const int k0 = K0 + 2 * v + j;          // khalf = 0
                        const int k1 = k0 + 16;                 // khalf = 1
                        const int o0 = (k0 / 25) * PSZ + ((k0 % 25) / 5) * PW + (k0 % 5);
                        const int o1 = (k1 / 25) * PSZ + ((k1 % 25) / 5) * PW + (k1 % 5);
                        bf[2 * v + j] = zxp[(khalf ? o1 : o0) + base];
                    }
                }
                acc = __builtin_amdgcn_wmma_f32_16x16x32_f16(
                          false, af, false, bf, (short)0, acc, false, false);
            }
            if (lane < 16) {                        // rows 0..5 valid (lanes 0-15)
                const int hb = (py + 2) * PW + px + 2;
                #pragma unroll
                for (int r = 0; r < 6; ++r) {
                    float v = leaky_(acc[r] + sb1[r]);
                    hpad[r * PSZ + hb] = (_Float16)v;
                    ls[r >> 1] += v;
                    lq[r >> 1] += v * v;
                }
            }
        }
        // wave-level butterfly reduce + one atomic per wave per group
        #pragma unroll
        for (int g = 0; g < 3; ++g) {
            float s = ls[g], q = lq[g];
            #pragma unroll
            for (int off = 16; off > 0; off >>= 1) {
                s += __shfl_xor(s, off, 32);
                q += __shfl_xor(q, off, 32);
            }
            if (lane == 0) { atomicAdd(&gsum[g], s); atomicAdd(&gsq[g], q); }
        }
        __syncthreads();

        // ---------- GroupNorm (3 groups x 2 ch x 1024 px = 2048 elems/group) ----------
        float mu[3], rs[3];
        #pragma unroll
        for (int g = 0; g < 3; ++g) {
            const float m_ = gsum[g] * (1.0f / 2048.0f);
            const float v_ = gsq[g] * (1.0f / 2048.0f) - m_ * m_;
            mu[g] = m_;
            rs[g] = rsqrtf(v_ + 1e-5f);
        }
        __syncthreads();                 // all gsum/gsq reads done
        if (tid < 3) { gsum[tid] = 0.0f; gsq[tid] = 0.0f; }
        for (int i = tid; i < 6 * 1024; i += 256) {
            const int c = i >> 10, p = i & 1023;
            const int idx = c * PSZ + ((p >> 5) + 2) * PW + (p & 31) + 2;
            float v = (float)hpad[idx];
            v = (v - mu[c >> 1]) * rs[c >> 1] * sg[c] + sbt[c];
            hpad[idx] = (_Float16)v;
        }
        __syncthreads();

        // ---------- conv2 (6->5, 5x5 pad 2) + leaky + damped update ----------
        for (int tt = 0; tt < 8; ++tt) {
            const int tile = wave * 8 + tt;
            const int p    = tile * 16 + nCol;
            const int py   = p >> 5, px = p & 31;
            const int base = py * PW + px;
            v8f acc = {};
            #pragma unroll
            for (int ch = 0; ch < 5; ++ch) {        // K chunks of 32 (160 total)
                const int K0 = ch * 32;
                v16h af, bf;
                #pragma unroll
                for (int v = 0; v < 8; ++v) {
                    const int kk = K0 + khalf * 16 + 2 * v;
                    const h2 pr = *(const h2*)&wa2[nCol][kk];
                    af[2 * v]     = pr.x;
                    af[2 * v + 1] = pr.y;
                }
                #pragma unroll
                for (int v = 0; v < 8; ++v) {
                    #pragma unroll
                    for (int j = 0; j < 2; ++j) {
                        const int k0 = K0 + 2 * v + j;
                        const int k1 = k0 + 16;
                        const int o0 = (k0 / 25) * PSZ + ((k0 % 25) / 5) * PW + (k0 % 5);
                        const int o1 = (k1 / 25) * PSZ + ((k1 % 25) / 5) * PW + (k1 % 5);
                        bf[2 * v + j] = hpad[(khalf ? o1 : o0) + base];
                    }
                }
                acc = __builtin_amdgcn_wmma_f32_16x16x32_f16(
                          false, af, false, bf, (short)0, acc, false, false);
            }
            if (lane < 16) {                        // z = 0.5 z + 0.5 leaky(conv2)
                const int zb = (py + 2) * PW + px + 2;
                #pragma unroll
                for (int r = 0; r < 5; ++r) {
                    const float v   = leaky_(acc[r] + sb2[r]);
                    const float old = (float)zxp[r * PSZ + zb];
                    zxp[r * PSZ + zb] = (_Float16)(0.5f * old + 0.5f * v);
                }
            }
        }
        __syncthreads();                 // zxp ready for next iteration's conv1
    }

    // ===== detection head: out[o] = sum_{c,p} z[c][p]*wh[o][c][p] + bh[o] =====
    if (tid < 10) oacc[tid] = 0.0f;
    __syncthreads();

    float po[10];
    #pragma unroll
    for (int o = 0; o < 10; ++o) po[o] = 0.0f;
    for (int p = tid; p < 1024; p += 256) {
        __builtin_prefetch(&wh[p], 0, 0);           // global_prefetch into L2/L0
        const int idx = ((p >> 5) + 2) * PW + (p & 31) + 2;
        float zv[5];
        #pragma unroll
        for (int c = 0; c < 5; ++c) zv[c] = (float)zxp[c * PSZ + idx];
        #pragma unroll
        for (int o = 0; o < 10; ++o) {
            float s = 0.0f;
            #pragma unroll
            for (int c = 0; c < 5; ++c)
                s += zv[c] * wh[(o * 5 + c) * 1024 + p];
            po[o] += s;
        }
    }
    #pragma unroll
    for (int o = 0; o < 10; ++o) {
        float s = po[o];
        #pragma unroll
        for (int off = 16; off > 0; off >>= 1) s += __shfl_xor(s, off, 32);
        if (lane == 0) atomicAdd(&oacc[o], s);
    }
    __syncthreads();
    if (tid < 10) out[(size_t)img * 10 + tid] = oacc[tid] + bh[tid];
}

extern "C" void kernel_launch(void* const* d_in, const int* in_sizes, int n_in,
                              void* d_out, int out_size, void* d_ws, size_t ws_size,
                              hipStream_t stream) {
    const float* image = (const float*)d_in[0];
    const float* w1    = (const float*)d_in[1];
    const float* b1    = (const float*)d_in[2];
    const float* gamma = (const float*)d_in[3];
    const float* beta  = (const float*)d_in[4];
    const float* w2    = (const float*)d_in[5];
    const float* b2    = (const float*)d_in[6];
    const float* wh    = (const float*)d_in[7];
    const float* bh    = (const float*)d_in[8];
    float* out = (float*)d_out;

    const int nImg = in_sizes[0] / (3 * 32 * 32);   // 1024
    deq_fused_kernel<<<nImg, 256, 0, stream>>>(image, w1, b1, gamma, beta,
                                               w2, b2, wh, bh, out);
}
